// DecoderMultiHeadSelfAttention_49065706389550
// MI455X (gfx1250) — compile-verified, matrix operand order checked by reading, not run on hardware
//
#include <hip/hip_runtime.h>
#include <hip/hip_bf16.h>
#include <stdint.h>

// ---------------------------------------------------------------------------
// CDNA5 (gfx1250) decoder MHA with KV cache.
// Matrix math: v_wmma_f32_16x16x32_bf16, fp32 accumulate.
// Fragment layouts per CDNA5 ISA §7.12.2:
//   A (16x32, MxK): lane L -> row M=L&15; elem e<8:  K = hi*8+e
//                                          elem e>=8: K = 16+hi*8+(e-8)   (hi = L>>4)
//   B (32x16, KxN): lane L -> col N=L&15; same K mapping per element.
//   C/D (16x16 f32): lane L -> col N=L&15; VGPR r -> row M = hi*8 + r.
// ---------------------------------------------------------------------------

typedef __attribute__((ext_vector_type(16))) __bf16 bf16x16;
typedef __attribute__((ext_vector_type(8)))  float  f32x8;

union BFrag {
  bf16x16 v;
  unsigned short s[16];
  uint4 q[2];
};

__device__ __forceinline__ unsigned short f32_bf16(float f) {
  unsigned int u = __float_as_uint(f);
  u += 0x7FFFu + ((u >> 16) & 1u);     // round-to-nearest-even
  return (unsigned short)(u >> 16);
}

// Load one WMMA fragment (16 bf16) from a row-major bf16 row pointer.
__device__ __forceinline__ bf16x16 load_bf_frag(const unsigned short* __restrict__ row,
                                                int k0, int hi) {
  BFrag f;
  f.q[0] = *reinterpret_cast<const uint4*>(row + k0 + hi * 8);        // K run 0 (8 bf16)
  f.q[1] = *reinterpret_cast<const uint4*>(row + k0 + 16 + hi * 8);   // K run 1 (8 bf16)
  return f.v;
}

// Build a fragment from an f32 row (convert in-register).
__device__ __forceinline__ bf16x16 cvt_frag_f32(const float* __restrict__ row,
                                                int k0, int hi) {
  BFrag f;
#pragma unroll
  for (int g = 0; g < 2; ++g) {
    const float4* p = reinterpret_cast<const float4*>(row + k0 + g * 16 + hi * 8);
    float4 a = p[0];
    float4 b = p[1];
    f.s[g * 8 + 0] = f32_bf16(a.x); f.s[g * 8 + 1] = f32_bf16(a.y);
    f.s[g * 8 + 2] = f32_bf16(a.z); f.s[g * 8 + 3] = f32_bf16(a.w);
    f.s[g * 8 + 4] = f32_bf16(b.x); f.s[g * 8 + 5] = f32_bf16(b.y);
    f.s[g * 8 + 6] = f32_bf16(b.z); f.s[g * 8 + 7] = f32_bf16(b.w);
  }
  return f.v;
}

// ---------------------------------------------------------------------------
// fp32 -> bf16 bulk convert (4 elems / thread, grid-stride)
// ---------------------------------------------------------------------------
__global__ void mha_cvt_bf16(const float* __restrict__ in,
                             unsigned short* __restrict__ out, int n4) {
  int i = blockIdx.x * blockDim.x + threadIdx.x;
  int stride = gridDim.x * blockDim.x;
  for (; i < n4; i += stride) {
    float4 v = reinterpret_cast<const float4*>(in)[i];
    uint2 p;
    p.x = (unsigned int)f32_bf16(v.x) | ((unsigned int)f32_bf16(v.y) << 16);
    p.y = (unsigned int)f32_bf16(v.z) | ((unsigned int)f32_bf16(v.w) << 16);
    reinterpret_cast<uint2*>(out)[i] = p;
  }
}

// ---------------------------------------------------------------------------
// C(MxN,f32) = A(MxK,bf16) @ W(NxK,bf16)^T + bias   (i.e. x @ W.T + b)
// One wave -> 16x64 output tile (4 accumulators), K-loop step 32.
// ---------------------------------------------------------------------------
__global__ __launch_bounds__(256)
void mha_gemm_bf16(const unsigned short* __restrict__ A,
                   const unsigned short* __restrict__ W,
                   const float* __restrict__ bias,
                   float* __restrict__ C,
                   int M, int N, int K) {
  int wave  = (blockIdx.x * blockDim.x + threadIdx.x) >> 5;
  int lane  = threadIdx.x & 31;
  int ngrps = N >> 6;                       // groups of 64 columns
  int m_tile = wave / ngrps;
  int n_grp  = wave % ngrps;
  if (m_tile * 16 >= M) return;
  int hi = lane >> 4;
  int lq = lane & 15;

  const unsigned short* arow = A + (size_t)(m_tile * 16 + lq) * K;
  const unsigned short* wrow[4];
#pragma unroll
  for (int t = 0; t < 4; ++t)
    wrow[t] = W + (size_t)(n_grp * 64 + t * 16 + lq) * K;

  f32x8 acc[4];
#pragma unroll
  for (int t = 0; t < 4; ++t) acc[t] = (f32x8)(0.0f);

  for (int k0 = 0; k0 < K; k0 += 32) {
    bf16x16 af = load_bf_frag(arow, k0, hi);
#pragma unroll
    for (int t = 0; t < 4; ++t) {
      bf16x16 bf = load_bf_frag(wrow[t], k0, hi);
      acc[t] = __builtin_amdgcn_wmma_f32_16x16x32_bf16(
          false, af, false, bf, (short)0, acc[t], false, false);
    }
  }

#pragma unroll
  for (int t = 0; t < 4; ++t) {
    int n = n_grp * 64 + t * 16 + lq;
    float bv = bias[n];
#pragma unroll
    for (int r = 0; r < 8; ++r) {
      int m = m_tile * 16 + hi * 8 + r;
      C[(size_t)m * N + n] = acc[t][r] + bv;
    }
  }
}

// ---------------------------------------------------------------------------
// KV-cache copy + update of last q_len rows. float4 traffic, grid-stride.
// Per cache: 32*2048*2048 f32 = 33,554,432 float4.
// ---------------------------------------------------------------------------
__global__ void mha_cache_update(const float4* __restrict__ kc,
                                 const float4* __restrict__ vc,
                                 const float4* __restrict__ knew,
                                 const float4* __restrict__ vnew,
                                 float4* __restrict__ kout,
                                 float4* __restrict__ vout) {
  const long long total = 33554432LL;
  long long i = (long long)blockIdx.x * blockDim.x + threadIdx.x;
  long long stride = (long long)gridDim.x * blockDim.x;
  for (; i < total; i += stride) {
    int s = (int)((i >> 9) & 2047);            // seq position (2048/4=512 float4 per row)
    if (s >= 2032) {                           // last q_len=16 rows come from new K/V
      long long b  = i >> 20;
      long long d4 = i & 511;
      long long src = ((b * 16 + (s - 2032)) << 9) + d4;
      kout[i] = knew[src];
      vout[i] = vnew[src];
    } else {
      kout[i] = kc[i];
      vout[i] = vc[i];
    }
  }
}

// ---------------------------------------------------------------------------
// Flash attention, one wave per (batch, head).
// S^T = K_chunk(32x128) @ q^T(128x16) -> softmax stats lane-local (+1 shfl_xor),
// P transposed to A-layout via per-wave LDS, then O += P(16x32) @ V(32x128).
// ---------------------------------------------------------------------------
__global__ __launch_bounds__(256)
void mha_attention(const float* __restrict__ qf,      // (32,16,2048) f32
                   const float* __restrict__ kcache,  // (32,2048,2048) updated
                   const float* __restrict__ vcache,
                   const int* __restrict__ mask,      // (32,16,2048)
                   float* __restrict__ attn_out) {    // (32,16,2048)
  __shared__ float plds[8][16 * 36];                  // per-wave padded 16x32 P tile
  int wslot = threadIdx.x >> 5;
  int lane  = threadIdx.x & 31;
  int job = blockIdx.x * 8 + wslot;                   // 512 jobs = b*16 + h
  int b = job >> 4;
  int h = job & 15;
  int hi = lane >> 4;
  int lq = lane & 15;
  float* pw = &plds[wslot][0];

  const float scale = 0.08838834764831845f;           // 1/sqrt(128)

  // q fragments (q row = lq), reused as B of K@q^T for all chunks
  const float* qrow = qf + (size_t)(b * 16 + lq) * 2048 + h * 128;
  bf16x16 qfr[4];
#pragma unroll
  for (int ks = 0; ks < 4; ++ks) qfr[ks] = cvt_frag_f32(qrow, ks * 32, hi);

  const float* kbase = kcache + (size_t)b * 2048 * 2048 + h * 128;
  const float* vbase = vcache + (size_t)b * 2048 * 2048 + h * 128;
  const int* mbase = mask + (size_t)(b * 16 + lq) * 2048;

  f32x8 O[8];
#pragma unroll
  for (int t = 0; t < 8; ++t) O[t] = (f32x8)(0.0f);
  float mrow = -3.0e38f;
  float lrow = 0.0f;

  for (int kv0 = 0; kv0 < 2048; kv0 += 32) {
    // ---- S^T tiles: st[t][r] = scores[q=lq][kv = kv0 + t*16 + hi*8 + r] ----
    f32x8 st[2];
#pragma unroll
    for (int t = 0; t < 2; ++t) {
      f32x8 s = (f32x8)(0.0f);
      const float* krow = kbase + (size_t)(kv0 + t * 16 + lq) * 2048;
#pragma unroll
      for (int ks = 0; ks < 4; ++ks) {
        bf16x16 kf = cvt_frag_f32(krow, ks * 32, hi);
        s = __builtin_amdgcn_wmma_f32_16x16x32_bf16(
            false, kf, false, qfr[ks], (short)0, s, false, false);
      }
      st[t] = s;
    }

    // ---- scale + mask + online softmax stats ----
    float vals[16];
    float cmax = -3.0e38f;
#pragma unroll
    for (int t = 0; t < 2; ++t)
#pragma unroll
      for (int r = 0; r < 8; ++r) {
        int kv = kv0 + t * 16 + hi * 8 + r;
        float s = st[t][r] * scale;
        if (mbase[kv] == 0) s = -1.0e9f;
        vals[t * 8 + r] = s;
        cmax = fmaxf(cmax, s);
      }
    cmax = fmaxf(cmax, __shfl_xor(cmax, 16, 32));
    float mnew = fmaxf(mrow, cmax);
    float c = __expf(mrow - mnew);
    float psum = 0.0f;
#pragma unroll
    for (int i = 0; i < 16; ++i) {
      float p = __expf(vals[i] - mnew);
      vals[i] = p;
      psum += p;
    }
    psum += __shfl_xor(psum, 16, 32);
    lrow = lrow * c + psum;
    mrow = mnew;

    // ---- transpose P to A-layout through LDS (per-wave private region) ----
    __syncthreads();
#pragma unroll
    for (int t = 0; t < 2; ++t)
#pragma unroll
      for (int r = 0; r < 8; ++r)
        pw[lq * 36 + t * 16 + hi * 8 + r] = vals[t * 8 + r];
    __syncthreads();

    BFrag pa;
#pragma unroll
    for (int g = 0; g < 2; ++g) {
      const float4* p0 = reinterpret_cast<const float4*>(pw + lq * 36 + g * 16 + hi * 8);
      float4 a = p0[0];
      float4 bb = p0[1];
      pa.s[g * 8 + 0] = f32_bf16(a.x);  pa.s[g * 8 + 1] = f32_bf16(a.y);
      pa.s[g * 8 + 2] = f32_bf16(a.z);  pa.s[g * 8 + 3] = f32_bf16(a.w);
      pa.s[g * 8 + 4] = f32_bf16(bb.x); pa.s[g * 8 + 5] = f32_bf16(bb.y);
      pa.s[g * 8 + 6] = f32_bf16(bb.z); pa.s[g * 8 + 7] = f32_bf16(bb.w);
    }

    // ---- rescale O rows by c (row m = hi*8+r lives on lane m) ----
    float crow[8];
#pragma unroll
    for (int r = 0; r < 8; ++r) crow[r] = __shfl(c, hi * 8 + r, 32);
#pragma unroll
    for (int t = 0; t < 8; ++t)
#pragma unroll
      for (int r = 0; r < 8; ++r) O[t][r] *= crow[r];

    // ---- O += P(16x32) @ V_chunk(32x128): 8 dk tiles ----
#pragma unroll
    for (int nt = 0; nt < 8; ++nt) {
      BFrag vb;
#pragma unroll
      for (int g = 0; g < 2; ++g)
#pragma unroll
        for (int w = 0; w < 8; ++w) {
          int kv = kv0 + g * 16 + hi * 8 + w;
          vb.s[g * 8 + w] = f32_bf16(vbase[(size_t)kv * 2048 + nt * 16 + lq]);
        }
      O[nt] = __builtin_amdgcn_wmma_f32_16x16x32_bf16(
          false, pa.v, false, vb.v, (short)0, O[nt], false, false);
    }
  }

  // ---- normalize and store: attn_out[b][m][h*128 + nt*16 + lq] ----
  float linv = 1.0f / lrow;
  float lr[8];
#pragma unroll
  for (int r = 0; r < 8; ++r) lr[r] = __shfl(linv, hi * 8 + r, 32);
#pragma unroll
  for (int nt = 0; nt < 8; ++nt)
#pragma unroll
    for (int r = 0; r < 8; ++r) {
      int m = hi * 8 + r;
      attn_out[(size_t)(b * 16 + m) * 2048 + h * 128 + nt * 16 + lq] = O[nt][r] * lr[r];
    }
}

// ---------------------------------------------------------------------------
// Host launch
// ---------------------------------------------------------------------------
extern "C" void kernel_launch(void* const* d_in, const int* in_sizes, int n_in,
                              void* d_out, int out_size, void* d_ws, size_t ws_size,
                              hipStream_t stream) {
  const float* x  = (const float*)d_in[0];
  const float* kc = (const float*)d_in[1];
  const float* vc = (const float*)d_in[2];
  const int*   mk = (const int*)d_in[3];
  const float* Wq = (const float*)d_in[4];
  const float* bq = (const float*)d_in[5];
  const float* Wk = (const float*)d_in[6];
  const float* bk = (const float*)d_in[7];
  const float* Wv = (const float*)d_in[8];
  const float* bv = (const float*)d_in[9];
  const float* Wo = (const float*)d_in[10];
  const float* bo = (const float*)d_in[11];

  const size_t X_N = 1048576;     // 32*16*2048
  const size_t W_N = 4194304;     // 2048*2048
  const size_t C_N = 134217728;   // 32*2048*2048

  // workspace layout (bytes)
  char* ws = (char*)d_ws;
  unsigned short* x_bf    = (unsigned short*)(ws);
  unsigned short* Wq_bf   = (unsigned short*)(ws + 2097152);
  unsigned short* Wk_bf   = (unsigned short*)(ws + 10485760);
  unsigned short* Wv_bf   = (unsigned short*)(ws + 18874368);
  unsigned short* Wo_bf   = (unsigned short*)(ws + 27262976);
  float*          q_f32   = (float*)(ws + 35651584);
  float*          k_f32   = (float*)(ws + 39845888);
  float*          v_f32   = (float*)(ws + 44040192);
  float*          attn_f  = (float*)(ws + 48234496);
  unsigned short* attn_bf = (unsigned short*)(ws + 52428800);
  // total <= 54,525,952 bytes (~52 MB)

  float* out_p = (float*)d_out;
  float* k_out = out_p + X_N;          // updated k_cache output
  float* v_out = k_out + C_N;          // updated v_cache output

  // 1) converts to bf16
  mha_cvt_bf16<<<1024, 256, 0, stream>>>(x,  x_bf,  (int)(X_N / 4));
  mha_cvt_bf16<<<2048, 256, 0, stream>>>(Wq, Wq_bf, (int)(W_N / 4));
  mha_cvt_bf16<<<2048, 256, 0, stream>>>(Wk, Wk_bf, (int)(W_N / 4));
  mha_cvt_bf16<<<2048, 256, 0, stream>>>(Wv, Wv_bf, (int)(W_N / 4));
  mha_cvt_bf16<<<2048, 256, 0, stream>>>(Wo, Wo_bf, (int)(W_N / 4));

  // 2) QKV projections: (512x2048) @ (2048x2048)^T ; 32*32 wave jobs = 128 blocks
  mha_gemm_bf16<<<128, 256, 0, stream>>>(x_bf, Wq_bf, bq, q_f32, 512, 2048, 2048);
  mha_gemm_bf16<<<128, 256, 0, stream>>>(x_bf, Wk_bf, bk, k_f32, 512, 2048, 2048);
  mha_gemm_bf16<<<128, 256, 0, stream>>>(x_bf, Wv_bf, bv, v_f32, 512, 2048, 2048);

  // 3) KV cache copy + last-16-row update (bulk HBM traffic)
  mha_cache_update<<<4096, 256, 0, stream>>>(
      (const float4*)kc, (const float4*)vc,
      (const float4*)k_f32, (const float4*)v_f32,
      (float4*)k_out, (float4*)v_out);

  // 4) flash attention over updated caches: 512 (b,h) wave jobs = 64 blocks
  mha_attention<<<64, 256, 0, stream>>>(q_f32, k_out, v_out, mk, attn_f);

  // 5) output projection
  mha_cvt_bf16<<<1024, 256, 0, stream>>>(attn_f, attn_bf, (int)(X_N / 4));
  mha_gemm_bf16<<<128, 256, 0, stream>>>(attn_bf, Wo_bf, bo, out_p, 512, 2048, 2048);
}
